// Transformer_74191265071150
// MI455X (gfx1250) — compile-verified, matrix operand order checked by reading, not run on hardware
//
#include <hip/hip_runtime.h>
#include <hip/hip_bf16.h>

// ---------------------------------------------------------------------------
// CDNA5 (gfx1250) transformer block: WMMA f16 GEMMs (register-blocked 32x64
// per wave) + flash attention. fp32 accumulate everywhere, fp32 LN/softmax.
// ---------------------------------------------------------------------------

typedef __attribute__((ext_vector_type(16))) _Float16 v16h;
typedef __attribute__((ext_vector_type(8)))  float    v8f;

#define N_TOK   1024
#define C_DIM   256
#define HEADS   8
#define DH      32
#define HID     512
#define BATCH   16
#define ROWS    (BATCH * N_TOK)            // 16384 token rows
#define QKV_LD  768
#define ATT_SCALE 0.17677669529663687f     // 1/sqrt(32)

// K-index mapping for 16-bit A/B fragments (ISA 7.12.2):
// VGPR j holds element pair K = 2*(j%4) + 16*(j/4) + 8*(lane/16) (+0,+1)
__device__ __forceinline__ int kmap(int j, int half) {
    return 2 * (j & 3) + 16 * (j >> 2) + 8 * half;
}

// A fragment: row-major A (M x K), tile at (0,0) of the passed pointer.
__device__ __forceinline__ v16h load_a_frag(const _Float16* __restrict__ A, int lda) {
    int lane = threadIdx.x & 31;
    int m = lane & 15, half = lane >> 4;
    v16h a;
#pragma unroll
    for (int j = 0; j < 8; ++j) {
        int k = kmap(j, half);
        const _Float16* p = A + (size_t)m * lda + k;   // K-pairs contiguous -> b128 loads
        a[2 * j]     = p[0];
        a[2 * j + 1] = p[1];
    }
    return a;
}

// B fragment from transposed storage Bt[n][k] (so K-pairs are contiguous).
__device__ __forceinline__ v16h load_b_frag_t(const _Float16* __restrict__ Bt, int ldb) {
    int lane = threadIdx.x & 31;
    int n = lane & 15, half = lane >> 4;
    v16h b;
#pragma unroll
    for (int j = 0; j < 8; ++j) {
        int k = kmap(j, half);
        const _Float16* p = Bt + (size_t)n * ldb + k;
        b[2 * j]     = p[0];
        b[2 * j + 1] = p[1];
    }
    return b;
}

// B fragment from row-major B[k][n] (used for the V matrix in attention).
__device__ __forceinline__ v16h load_b_frag_rm(const _Float16* __restrict__ B, int ldb) {
    int lane = threadIdx.x & 31;
    int n = lane & 15, half = lane >> 4;
    v16h b;
#pragma unroll
    for (int j = 0; j < 8; ++j) {
        int k = kmap(j, half);
        b[2 * j]     = B[(size_t)k * ldb + n];
        b[2 * j + 1] = B[(size_t)(k + 1) * ldb + n];
    }
    return b;
}

// ---------------------------------------------------------------------------
// Register-blocked GEMM core: one wave computes a 32x64 macro-tile
// (2 A-frags x 4 B-frags -> 8 wmmas per 6 fragment loads per k-step).
// ---------------------------------------------------------------------------
template <int KTILES>
__device__ __forceinline__ void gemm_block(const _Float16* __restrict__ A, int lda,
                                           const _Float16* __restrict__ Bt, int ldb,
                                           int mt0, int nt0, v8f acc[2][4]) {
    const _Float16* A0 = A + (size_t)(mt0 + 0) * 16 * lda;
    const _Float16* A1 = A + (size_t)(mt0 + 1) * 16 * lda;
#pragma unroll
    for (int kt = 0; kt < KTILES; ++kt) {
        __builtin_prefetch(A0 + (kt + 1) * 32, 0, 1);   // global_prefetch_b8 on A stream
        __builtin_prefetch(A1 + (kt + 1) * 32, 0, 1);
        v16h a0 = load_a_frag(A0 + kt * 32, lda);
        v16h a1 = load_a_frag(A1 + kt * 32, lda);
#pragma unroll
        for (int j = 0; j < 4; ++j) {
            v16h b = load_b_frag_t(Bt + (size_t)(nt0 + j) * 16 * ldb + kt * 32, ldb);
            acc[0][j] = __builtin_amdgcn_wmma_f32_16x16x32_f16(false, a0, false, b,
                                                               (short)0, acc[0][j], false, false);
            acc[1][j] = __builtin_amdgcn_wmma_f32_16x16x32_f16(false, a1, false, b,
                                                               (short)0, acc[1][j], false, false);
        }
    }
}

// ---------------------------------------------------------------------------
// Weight transpose + f32 -> f16 convert: dst[n*K + k] = src[k*Nn + n]
// ---------------------------------------------------------------------------
__global__ void transpose_to_f16(const float* __restrict__ src, _Float16* __restrict__ dst,
                                 int K, int Nn) {
    int idx = blockIdx.x * 256 + threadIdx.x;
    if (idx >= K * Nn) return;
    int k = idx / Nn, n = idx % Nn;
    dst[(size_t)n * K + k] = (_Float16)src[idx];
}

// bias16[h][n][m] = table[rel_index[n*N+m], h]
__global__ void bias_precompute(const float* __restrict__ table, const int* __restrict__ rel,
                                _Float16* __restrict__ bias16) {
    int idx = blockIdx.x * 256 + threadIdx.x;     // n*N + m, < 2^20
    int ri = rel[idx];
#pragma unroll
    for (int h = 0; h < HEADS; ++h)
        bias16[((size_t)h << 20) + idx] = (_Float16)table[ri * HEADS + h];
}

// ---------------------------------------------------------------------------
// LayerNorm kernels (one block of 256 threads per token row)
// ---------------------------------------------------------------------------
__global__ __launch_bounds__(256) void ln1_transpose(const float* __restrict__ x,
                                                     const float* __restrict__ g,
                                                     const float* __restrict__ beta,
                                                     float* __restrict__ t,
                                                     _Float16* __restrict__ h16) {
    __shared__ float red[256];
    int bn = blockIdx.x;               // b*1024 + n
    int b = bn >> 10, n = bn & 1023;
    int c = threadIdx.x;
    float v = x[((size_t)b * C_DIM + c) * N_TOK + n];

    red[c] = v; __syncthreads();
    for (int s = 128; s > 0; s >>= 1) { if (c < s) red[c] += red[c + s]; __syncthreads(); }
    float mu = red[0] * (1.0f / C_DIM);
    __syncthreads();
    float d = v - mu;
    red[c] = d * d; __syncthreads();
    for (int s = 128; s > 0; s >>= 1) { if (c < s) red[c] += red[c + s]; __syncthreads(); }
    float var = red[0] * (1.0f / C_DIM);

    float y = d * rsqrtf(var + 1e-5f) * g[c] + beta[c];
    t[(size_t)bn * C_DIM + c]   = v;
    h16[(size_t)bn * C_DIM + c] = (_Float16)y;
}

__global__ __launch_bounds__(256) void ln2_rows(const float* __restrict__ t,
                                                const float* __restrict__ g,
                                                const float* __restrict__ beta,
                                                _Float16* __restrict__ h16) {
    __shared__ float red[256];
    int bn = blockIdx.x;
    int c = threadIdx.x;
    float v = t[(size_t)bn * C_DIM + c];

    red[c] = v; __syncthreads();
    for (int s = 128; s > 0; s >>= 1) { if (c < s) red[c] += red[c + s]; __syncthreads(); }
    float mu = red[0] * (1.0f / C_DIM);
    __syncthreads();
    float d = v - mu;
    red[c] = d * d; __syncthreads();
    for (int s = 128; s > 0; s >>= 1) { if (c < s) red[c] += red[c + s]; __syncthreads(); }
    float var = red[0] * (1.0f / C_DIM);

    h16[(size_t)bn * C_DIM + c] = (_Float16)(d * rsqrtf(var + 1e-5f) * g[c] + beta[c]);
}

// ---------------------------------------------------------------------------
// GEMM kernels: one wave per 32x64 macro-tile, 8 waves per block
// ---------------------------------------------------------------------------
__global__ __launch_bounds__(256) void gemm_qkv(const _Float16* __restrict__ H,
                                                const _Float16* __restrict__ Wt,
                                                _Float16* __restrict__ qkv) {
    int wave = threadIdx.x >> 5;
    int job = blockIdx.x * 8 + wave;
    const int NB = QKV_LD / 64;                     // 12
    int mt0 = (job / NB) * 2, nt0 = (job % NB) * 4;
    v8f acc[2][4] = {};
    gemm_block<8>(H, C_DIM, Wt, C_DIM, mt0, nt0, acc);

    int lane = threadIdx.x & 31, n = lane & 15, half = lane >> 4;
#pragma unroll
    for (int i = 0; i < 2; ++i)
#pragma unroll
        for (int j = 0; j < 4; ++j)
#pragma unroll
            for (int r = 0; r < 8; ++r) {
                int row = (mt0 + i) * 16 + r + 8 * half;
                qkv[(size_t)row * QKV_LD + (nt0 + j) * 16 + n] = (_Float16)acc[i][j][r];
            }
}

__global__ __launch_bounds__(256) void gemm_proj_residual(const _Float16* __restrict__ AO,
                                                          const _Float16* __restrict__ WoT,
                                                          const float* __restrict__ bo,
                                                          float* __restrict__ t) {
    int wave = threadIdx.x >> 5;
    int job = blockIdx.x * 8 + wave;
    const int NB = C_DIM / 64;                      // 4
    int mt0 = (job / NB) * 2, nt0 = (job % NB) * 4;
    v8f acc[2][4] = {};
    gemm_block<8>(AO, C_DIM, WoT, C_DIM, mt0, nt0, acc);

    int lane = threadIdx.x & 31, n = lane & 15, half = lane >> 4;
#pragma unroll
    for (int i = 0; i < 2; ++i)
#pragma unroll
        for (int j = 0; j < 4; ++j) {
            int col = (nt0 + j) * 16 + n;
            float bias = bo[col];
#pragma unroll
            for (int r = 0; r < 8; ++r) {
                int row = (mt0 + i) * 16 + r + 8 * half;
                t[(size_t)row * C_DIM + col] += acc[i][j][r] + bias;
            }
        }
}

__global__ __launch_bounds__(256) void gemm_mlp1(const _Float16* __restrict__ H2,
                                                 const _Float16* __restrict__ W1T,
                                                 const float* __restrict__ b1,
                                                 _Float16* __restrict__ G) {
    int wave = threadIdx.x >> 5;
    int job = blockIdx.x * 8 + wave;
    const int NB = HID / 64;                        // 8
    int mt0 = (job / NB) * 2, nt0 = (job % NB) * 4;
    v8f acc[2][4] = {};
    gemm_block<8>(H2, C_DIM, W1T, C_DIM, mt0, nt0, acc);

    int lane = threadIdx.x & 31, n = lane & 15, half = lane >> 4;
#pragma unroll
    for (int i = 0; i < 2; ++i)
#pragma unroll
        for (int j = 0; j < 4; ++j) {
            int col = (nt0 + j) * 16 + n;
            float bias = b1[col];
#pragma unroll
            for (int r = 0; r < 8; ++r) {
                int row = (mt0 + i) * 16 + r + 8 * half;
                float u = acc[i][j][r] + bias;
                float ge = 0.5f * u * (1.0f + erff(u * 0.70710678118f));   // exact GELU
                G[(size_t)row * HID + col] = (_Float16)ge;
            }
        }
}

__global__ __launch_bounds__(256) void gemm_mlp2_out(const _Float16* __restrict__ G,
                                                     const _Float16* __restrict__ W2T,
                                                     const float* __restrict__ b2,
                                                     const float* __restrict__ t,
                                                     float* __restrict__ out) {
    int wave = threadIdx.x >> 5;
    int job = blockIdx.x * 8 + wave;
    const int NB = C_DIM / 64;                      // 4
    int mt0 = (job / NB) * 2, nt0 = (job % NB) * 4;
    v8f acc[2][4] = {};
    gemm_block<16>(G, HID, W2T, HID, mt0, nt0, acc);   // K = 512

    int lane = threadIdx.x & 31, n = lane & 15, half = lane >> 4;
#pragma unroll
    for (int i = 0; i < 2; ++i)
#pragma unroll
        for (int j = 0; j < 4; ++j) {
            int col = (nt0 + j) * 16 + n;
            float bias = b2[col];
#pragma unroll
            for (int r = 0; r < 8; ++r) {
                int row = (mt0 + i) * 16 + r + 8 * half;      // token row
                float val = t[(size_t)row * C_DIM + col] + acc[i][j][r] + bias;
                int b = row >> 10, tok = row & 1023;
                out[((size_t)b * C_DIM + col) * N_TOK + tok] = val;   // back to [B,C,H,W]
            }
        }
}

// ---------------------------------------------------------------------------
// Flash attention: one wave per 16-row q tile; 8 waves/block; grid = B*H*8
// ---------------------------------------------------------------------------
__global__ __launch_bounds__(256) void attention_flash(const _Float16* __restrict__ qkv,
                                                       const _Float16* __restrict__ bias,
                                                       _Float16* __restrict__ out) {
    __shared__ _Float16 plds[8][16][32];   // per-wave P tile (C-layout -> A-layout relay)

    int wave = threadIdx.x >> 5;
    int lane = threadIdx.x & 31;
    int n = lane & 15, half = lane >> 4;

    int bh = blockIdx.x >> 3;
    int b = bh >> 3, h = bh & 7;
    int qt = (blockIdx.x & 7) * 8 + wave;              // 0..63

    const _Float16* qbase = qkv + ((size_t)(b * N_TOK + qt * 16)) * QKV_LD + h * DH;
    v16h qf = load_a_frag(qbase, QKV_LD);              // Q tile 16x32 (K == DH)

    v8f o0 = {}, o1 = {};
    float mrun[8], lrun[8];
#pragma unroll
    for (int r = 0; r < 8; ++r) { mrun[r] = -3.0e38f; lrun[r] = 0.0f; }

    const _Float16* bias_h = bias + ((size_t)h << 20);

    for (int kt2 = 0; kt2 < 32; ++kt2) {               // 32 key tokens per iteration
        int ka = kt2 * 2, kb = ka + 1;

        // ---- scores: Q x K^T (two 16-key tiles, K-dim = DH = 32) ----
        const _Float16* kbA = qkv + ((size_t)(b * N_TOK + ka * 16)) * QKV_LD + C_DIM + h * DH;
        const _Float16* kbB = qkv + ((size_t)(b * N_TOK + kb * 16)) * QKV_LD + C_DIM + h * DH;
        v16h kfa = load_b_frag_t(kbA, QKV_LD);
        v16h kfb = load_b_frag_t(kbB, QKV_LD);
        v8f z = {};
        v8f da = __builtin_amdgcn_wmma_f32_16x16x32_f16(false, qf, false, kfa, (short)0, z, false, false);
        v8f db = __builtin_amdgcn_wmma_f32_16x16x32_f16(false, qf, false, kfb, (short)0, z, false, false);

        float sa[8], sb[8], mt[8];
#pragma unroll
        for (int r = 0; r < 8; ++r) {
            int qrow = qt * 16 + r + 8 * half;
            sa[r] = da[r] * ATT_SCALE + (float)bias_h[(size_t)qrow * N_TOK + ka * 16 + n];
            sb[r] = db[r] * ATT_SCALE + (float)bias_h[(size_t)qrow * N_TOK + kb * 16 + n];
            mt[r] = fmaxf(sa[r], sb[r]);
        }
        // row max across the 16 lanes holding this row (xor<=8 stays in half-group)
#pragma unroll
        for (int r = 0; r < 8; ++r) {
            mt[r] = fmaxf(mt[r], __shfl_xor(mt[r], 1, 32));
            mt[r] = fmaxf(mt[r], __shfl_xor(mt[r], 2, 32));
            mt[r] = fmaxf(mt[r], __shfl_xor(mt[r], 4, 32));
            mt[r] = fmaxf(mt[r], __shfl_xor(mt[r], 8, 32));
        }
        float pa[8], pb[8];
#pragma unroll
        for (int r = 0; r < 8; ++r) {
            float mnew = fmaxf(mrun[r], mt[r]);
            float alpha = __expf(mrun[r] - mnew);
            pa[r] = __expf(sa[r] - mnew);
            pb[r] = __expf(sb[r] - mnew);
            float rs = pa[r] + pb[r];
            rs += __shfl_xor(rs, 1, 32);
            rs += __shfl_xor(rs, 2, 32);
            rs += __shfl_xor(rs, 4, 32);
            rs += __shfl_xor(rs, 8, 32);
            lrun[r] = lrun[r] * alpha + rs;
            mrun[r] = mnew;
            o0[r] *= alpha;
            o1[r] *= alpha;
        }

        // ---- P tile: C-layout -> LDS -> A-layout fragment ----
#pragma unroll
        for (int r = 0; r < 8; ++r) {
            int qr = r + 8 * half;
            plds[wave][qr][n]      = (_Float16)pa[r];
            plds[wave][qr][16 + n] = (_Float16)pb[r];
        }
        v16h pf;
#pragma unroll
        for (int j = 0; j < 8; ++j) {
            int k = kmap(j, half);
            pf[2 * j]     = plds[wave][n][k];
            pf[2 * j + 1] = plds[wave][n][k + 1];
        }

        // ---- O += P x V (K-dim = 32 key tokens) ----
        const _Float16* vb = qkv + ((size_t)(b * N_TOK + kt2 * 32)) * QKV_LD + 2 * C_DIM + h * DH;
        v16h v0 = load_b_frag_rm(vb, QKV_LD);
        v16h v1 = load_b_frag_rm(vb + 16, QKV_LD);
        o0 = __builtin_amdgcn_wmma_f32_16x16x32_f16(false, pf, false, v0, (short)0, o0, false, false);
        o1 = __builtin_amdgcn_wmma_f32_16x16x32_f16(false, pf, false, v1, (short)0, o1, false, false);
    }

#pragma unroll
    for (int r = 0; r < 8; ++r) {
        int row = b * N_TOK + qt * 16 + r + 8 * half;
        float inv = 1.0f / lrun[r];
        out[(size_t)row * C_DIM + h * DH + n]      = (_Float16)(o0[r] * inv);
        out[(size_t)row * C_DIM + h * DH + 16 + n] = (_Float16)(o1[r] * inv);
    }
}

// ---------------------------------------------------------------------------
// Workspace layout (bytes)
// ---------------------------------------------------------------------------
constexpr size_t OFF_T     = 0;                                        // f32 residual [16384,256]
constexpr size_t OFF_H16   = OFF_T     + (size_t)ROWS * C_DIM * 4;     // f16 LN1 out
constexpr size_t OFF_WQKVT = OFF_H16   + (size_t)ROWS * C_DIM * 2;     // f16 Wqkv^T [768,256]
constexpr size_t OFF_WOT   = OFF_WQKVT + (size_t)QKV_LD * C_DIM * 2;   // f16 Wo^T [256,256]
constexpr size_t OFF_W1T   = OFF_WOT   + (size_t)C_DIM * C_DIM * 2;    // f16 W1^T [512,256]
constexpr size_t OFF_W2T   = OFF_W1T   + (size_t)HID * C_DIM * 2;      // f16 W2^T [256,512]
constexpr size_t OFF_QKV   = OFF_W2T   + (size_t)C_DIM * HID * 2;      // f16 qkv [16384,768]
constexpr size_t OFF_BIAS  = OFF_QKV   + (size_t)ROWS * QKV_LD * 2;    // f16 bias [8,1024,1024]
constexpr size_t OFF_AO    = OFF_BIAS  + (size_t)HEADS * N_TOK * N_TOK * 2; // f16 attn out
constexpr size_t OFF_H2    = OFF_AO    + (size_t)ROWS * C_DIM * 2;     // f16 LN2 out
constexpr size_t OFF_G     = OFF_H2    + (size_t)ROWS * C_DIM * 2;     // f16 MLP hidden

extern "C" void kernel_launch(void* const* d_in, const int* in_sizes, int n_in,
                              void* d_out, int out_size, void* d_ws, size_t ws_size,
                              hipStream_t stream) {
    const float* x       = (const float*)d_in[0];
    const int*   rel_idx = (const int*)  d_in[1];
    const float* ln1_g   = (const float*)d_in[2];
    const float* ln1_b   = (const float*)d_in[3];
    const float* Wqkv    = (const float*)d_in[4];
    const float* table   = (const float*)d_in[5];
    const float* Wo      = (const float*)d_in[6];
    const float* bo      = (const float*)d_in[7];
    const float* ln2_g   = (const float*)d_in[8];
    const float* ln2_b   = (const float*)d_in[9];
    const float* W1      = (const float*)d_in[10];
    const float* b1      = (const float*)d_in[11];
    const float* W2      = (const float*)d_in[12];
    const float* b2      = (const float*)d_in[13];

    char* ws = (char*)d_ws;
    float*     t      = (float*)    (ws + OFF_T);
    _Float16*  h16    = (_Float16*) (ws + OFF_H16);
    _Float16*  wqkvt  = (_Float16*) (ws + OFF_WQKVT);
    _Float16*  wot    = (_Float16*) (ws + OFF_WOT);
    _Float16*  w1t    = (_Float16*) (ws + OFF_W1T);
    _Float16*  w2t    = (_Float16*) (ws + OFF_W2T);
    _Float16*  qkv16  = (_Float16*) (ws + OFF_QKV);
    _Float16*  bias16 = (_Float16*) (ws + OFF_BIAS);
    _Float16*  ao16   = (_Float16*) (ws + OFF_AO);
    _Float16*  h2_16  = (_Float16*) (ws + OFF_H2);
    _Float16*  g16    = (_Float16*) (ws + OFF_G);
    float*     out    = (float*)d_out;

    // weight prep (f32 -> f16, transposed for B-fragment contiguity)
    transpose_to_f16<<<768, 256, 0, stream>>>(Wqkv, wqkvt, C_DIM, QKV_LD);
    transpose_to_f16<<<256, 256, 0, stream>>>(Wo,   wot,   C_DIM, C_DIM);
    transpose_to_f16<<<512, 256, 0, stream>>>(W1,   w1t,   C_DIM, HID);
    transpose_to_f16<<<512, 256, 0, stream>>>(W2,   w2t,   HID,   C_DIM);
    bias_precompute<<<4096, 256, 0, stream>>>(table, rel_idx, bias16);

    // attention sub-block
    ln1_transpose<<<ROWS, 256, 0, stream>>>(x, ln1_g, ln1_b, t, h16);
    gemm_qkv<<<(ROWS / 32) * (QKV_LD / 64) / 8, 256, 0, stream>>>(h16, wqkvt, qkv16);        // 768
    attention_flash<<<BATCH * HEADS * 8, 256, 0, stream>>>(qkv16, bias16, ao16);             // 1024
    gemm_proj_residual<<<(ROWS / 32) * (C_DIM / 64) / 8, 256, 0, stream>>>(ao16, wot, bo, t); // 256

    // feed-forward sub-block
    ln2_rows<<<ROWS, 256, 0, stream>>>(t, ln2_g, ln2_b, h2_16);
    gemm_mlp1<<<(ROWS / 32) * (HID / 64) / 8, 256, 0, stream>>>(h2_16, w1t, b1, g16);        // 512
    gemm_mlp2_out<<<(ROWS / 32) * (C_DIM / 64) / 8, 256, 0, stream>>>(g16, w2t, b2, t, out); // 256
}
